// MoleculeModel_438086664317
// MI455X (gfx1250) — matrix operand order repaired and checked by reading, not compiled, not run on hardware
//
#include <hip/hip_runtime.h>

// out[b] = x_b^T Q x_b  ==  sum_n (x@Q)[b,n] * x[b,n]
// fp16-split emulated-fp32 GEMM on v_wmma_f32_16x16x32_f16 (CDNA5 / gfx1250, wave32).
// hi/lo f16 fragments interleaved in 16-dword blocks so all LDS accesses in the
// hot loop use immediate DS offsets (<64KB) and B fetches are 64B-contiguous.

typedef __attribute__((ext_vector_type(16))) _Float16 v16h;
typedef __attribute__((ext_vector_type(8)))  float    v8f;
typedef __attribute__((ext_vector_type(4)))  int      v4i;

#define NF      512      // NUM_FEATURES
#define BATCH   65536
#define BM      64       // rows per workgroup
#define THREADS 128      // 4 waves, one 16-row tile per wave

union HFrag { v16h h; v4i i[2]; };

static __device__ __forceinline__ unsigned int pack2h(_Float16 a, _Float16 b) {
    unsigned int ua = (unsigned int)__builtin_bit_cast(unsigned short, a);
    unsigned int ub = (unsigned int)__builtin_bit_cast(unsigned short, b);
    return ua | (ub << 16);
}

// ---------------------------------------------------------------------------
// Prep: convert Q (fp32 row-major) into f16 hi/lo, stored directly in the
// WMMA B-fragment layout.  Block of 16 dwords per (nt,kt,lane):
//   dwords [0..7] = hi pairs (VGPR v), dwords [8..15] = lo pairs.
// B layout (16-bit B 32x16): lanes 0-15 hold K=0..7,16..23, lanes 16-31 K=8..15,24..31;
// VGPR v holds the K pair with v = ((k&7)>>1)|((k>>4)<<2) (low half = even K); lane%16 = column N.
// ---------------------------------------------------------------------------
__global__ void prep_q_kernel(const float* __restrict__ Q,
                              unsigned int* __restrict__ wsB) {
    int t = blockIdx.x * blockDim.x + threadIdx.x;   // 512 * 256 threads
    if (t >= NF * (NF / 2)) return;
    int n  = t & (NF - 1);       // coalesced along n
    int k  = (t >> 9) << 1;      // even k of the pair
    float q0 = Q[(size_t)k * NF + n];
    float q1 = Q[(size_t)(k + 1) * NF + n];
    _Float16 h0 = (_Float16)q0; _Float16 l0 = (_Float16)(q0 - (float)h0);
    _Float16 h1 = (_Float16)q1; _Float16 l1 = (_Float16)(q1 - (float)h1);
    int nt = n >> 4, nn = n & 15;
    int kt = k >> 5, kk = k & 31;
    int lane = ((kk >> 3) & 1) * 16 + nn; // lsel*16 + N
    int v    = ((kk & 7) >> 1) | ((kk >> 4) << 2);
    unsigned int dw = (((unsigned)(nt * 16 + kt) * 32 + lane) * 16) + v;
    wsB[dw]     = pack2h(h0, h1);
    wsB[dw + 8] = pack2h(l0, l1);
}

// ---------------------------------------------------------------------------
// Main fused kernel: 64 rows / workgroup, 4 waves, 16-row tile per wave.
// ---------------------------------------------------------------------------
__global__ __launch_bounds__(THREADS)
void quadform_kernel(const float* __restrict__ x,
                     const unsigned int* __restrict__ wsB,
                     float* __restrict__ out) {
    // Interleaved hi/lo A fragments for the 64x512 x-block: 128 KB
    __shared__ unsigned int sA[BM * NF];   // 32768 dwords

    const int lane  = threadIdx.x & 31;
    const int rt    = threadIdx.x >> 5;          // wave id == row tile 0..3
    const int wgrow = blockIdx.x * BM;

    // ---- one-time staging: x block -> hi/lo f16 A-fragment layout --------
    // A layout (16-bit A 16x32): lanes 0-15 own K in {0..7}u{16..23},
    // lanes 16-31 own K in {8..15}u{24..31}; v = ((k&7)>>1) | ((k>>4)<<2).
    #pragma unroll 4
    for (int j = 0; j < (BM * NF / 2) / THREADS; ++j) {   // 128 iterations
        int p   = threadIdx.x + j * THREADS;              // k-pair index
        int row = p >> 8;                                  // 256 pairs per row
        int k   = (p & 255) << 1;
        const float* xr = x + (size_t)(wgrow + row) * NF + k;
        float a0 = xr[0], a1 = xr[1];                      // coalesced b64
        _Float16 h0 = (_Float16)a0; _Float16 l0 = (_Float16)(a0 - (float)h0);
        _Float16 h1 = (_Float16)a1; _Float16 l1 = (_Float16)(a1 - (float)h1);
        int kt   = k >> 5, kk = k & 31;
        int lsel = (kk >> 3) & 1;
        int v    = ((kk & 7) >> 1) | ((kk >> 4) << 2);
        int ln2  = lsel * 16 + (row & 15);
        int rtt  = row >> 4;
        unsigned int dw = (((unsigned)(rtt * 16 + kt) * 32 + ln2) * 16) + v;
        sA[dw]     = pack2h(h0, h1);
        sA[dw + 8] = pack2h(l0, l1);
    }
    __syncthreads();

    float part[8];
    #pragma unroll
    for (int r = 0; r < 8; ++r) part[r] = 0.f;

    for (int nt = 0; nt < NF / 16; ++nt) {            // 32 column tiles
        v8f acc = {};
        #pragma unroll 4
        for (int kt = 0; kt < NF / 32; ++kt) {        // 16 k-steps of 32
            unsigned int ab = ((unsigned)(rt * 16 + kt) * 32 + lane) * 16;
            unsigned int bb = ((unsigned)(nt * 16 + kt) * 32 + lane) * 16;
            HFrag Ah, Al, Bh, Bl;
            Ah.i[0] = *(const v4i*)&sA[ab];      Ah.i[1] = *(const v4i*)&sA[ab + 4];
            Al.i[0] = *(const v4i*)&sA[ab + 8];  Al.i[1] = *(const v4i*)&sA[ab + 12];
            Bh.i[0] = *(const v4i*)&wsB[bb];     Bh.i[1] = *(const v4i*)&wsB[bb + 4];
            Bl.i[0] = *(const v4i*)&wsB[bb + 8]; Bl.i[1] = *(const v4i*)&wsB[bb + 12];
            // y += xh*Qh + xh*Ql + xl*Qh   (fp32 accumulate)
            acc = __builtin_amdgcn_wmma_f32_16x16x32_f16(false, Ah.h, false, Bh.h,
                                                         (short)0, acc, false, false);
            acc = __builtin_amdgcn_wmma_f32_16x16x32_f16(false, Ah.h, false, Bl.h,
                                                         (short)0, acc, false, false);
            acc = __builtin_amdgcn_wmma_f32_16x16x32_f16(false, Al.h, false, Bh.h,
                                                         (short)0, acc, false, false);
        }
        // ---- epilogue: part[r] += y[m, col] * x[m, col] ------------------
        // C/D layout: VGPR r -> M=r (lanes 0-15), M=r+8 (lanes 16-31); N=lane%16.
        int col  = nt * 16 + (lane & 15);
        int ekt  = col >> 5, ekk = col & 31;
        int elsl = (ekk >> 3) & 1;
        int ev   = ((ekk & 7) >> 1) | ((ekk >> 4) << 2);
        int ehalf = ekk & 1;
        int mhi  = (lane >> 4) << 3;
        #pragma unroll
        for (int r = 0; r < 8; ++r) {
            int m   = r + mhi;
            int ln2 = elsl * 16 + m;
            unsigned int dwb = (((unsigned)(rt * 16 + ekt) * 32 + ln2) * 16) + ev;
            unsigned int hw = sA[dwb];
            unsigned int lw = sA[dwb + 8];
            unsigned short hb = (unsigned short)(ehalf ? (hw >> 16) : (hw & 0xffffu));
            unsigned short lb = (unsigned short)(ehalf ? (lw >> 16) : (lw & 0xffffu));
            float xv = (float)__builtin_bit_cast(_Float16, hb)
                     + (float)__builtin_bit_cast(_Float16, lb);
            part[r] += acc[r] * xv;
        }
    }

    // ---- reduce across the 16-lane N groups, one store per row ----------
    #pragma unroll
    for (int r = 0; r < 8; ++r) {
        float val = part[r];
        val += __shfl_xor(val, 1, 32);
        val += __shfl_xor(val, 2, 32);
        val += __shfl_xor(val, 4, 32);
        val += __shfl_xor(val, 8, 32);
        if ((lane & 15) == 0) {
            int row = wgrow + rt * 16 + r + ((lane >> 4) << 3);
            out[row] = val;
        }
    }
}

extern "C" void kernel_launch(void* const* d_in, const int* in_sizes, int n_in,
                              void* d_out, int out_size, void* d_ws, size_t ws_size,
                              hipStream_t stream) {
    const float* x = (const float*)d_in[0];   // [65536, 512]
    const float* Q = (const float*)d_in[1];   // [512, 512]
    float* out = (float*)d_out;               // [65536]
    // Workspace: Q hi/lo f16 fragments interleaved, 512*512 dwords = 1 MB
    unsigned int* wsB = (unsigned int*)d_ws;

    prep_q_kernel<<<(NF * (NF / 2)) / 256, 256, 0, stream>>>(Q, wsB);
    quadform_kernel<<<BATCH / BM, THREADS, 0, stream>>>(x, wsB, out);
}
